// SlotAttention_87978110091988
// MI455X (gfx1250) — matrix operand order repaired and checked by reading, not compiled, not run on hardware
//
#include <hip/hip_runtime.h>
#include <math.h>

typedef _Float16 h16;
typedef __attribute__((ext_vector_type(16))) _Float16 v16h;
typedef __attribute__((ext_vector_type(8)))  float    v8f;

#define WB 16
#define WN 16384
#define WD 64
#define WS 7
#define WH 128
#define N_ITERS 3
#define N_MESH 4
#define N_SH 5
#define LOG_S 1.9459101090932196f
#define CEPS 1e-8f

static __device__ inline v8f wmma16(v16h a, v16h b, v8f c) {
  return __builtin_amdgcn_wmma_f32_16x16x32_f16(false, a, false, b, (short)0, c, false, false);
}

// A-fragment loader: 16x32 f16 tile, row stride ld (halves). M = m (0..15), hi = lane>=16.
static __device__ inline v16h load_a16x32(const h16* src, int ld, int m, bool hi, int kofs) {
  v16h a;
#pragma unroll
  for (int r = 0; r < 8; ++r) {
    int k0 = (hi ? 16 : 0) + 2 * r + kofs;
    a[2 * r]     = src[m * ld + k0];
    a[2 * r + 1] = src[m * ld + k0 + 1];
  }
  return a;
}

static __device__ inline float blockReduce64(float v, float* red, int tid) {
  red[tid] = v; __syncthreads();
#pragma unroll
  for (int o = 32; o > 0; o >>= 1) { if (tid < o) red[tid] += red[tid + o]; __syncthreads(); }
  float r = red[0]; __syncthreads(); return r;
}

static __device__ inline float blockReduce256(float v, float* red, int tid) {
  red[tid] = v; __syncthreads();
#pragma unroll
  for (int o = 128; o > 0; o >>= 1) { if (tid < o) red[tid] += red[tid + o]; __syncthreads(); }
  float r = red[0]; __syncthreads(); return r;
}

// ---------------- K1: fused input LayerNorm + K/V projections (WMMA) + wi logits ----------------
__global__ void k_ln_proj(const float* __restrict__ x,
                          const float* __restrict__ ln_g, const float* __restrict__ ln_b,
                          const float* __restrict__ Wk, const float* __restrict__ Wv,
                          const float* __restrict__ wi_w, const float* __restrict__ wi_b,
                          h16* __restrict__ kn_h, h16* __restrict__ val_h,
                          float* __restrict__ logit_i) {
  __shared__ h16 lds[4][16 * 64];
  int wave = threadIdx.x >> 5;
  int lane = threadIdx.x & 31;
  bool hi = lane >= 16;
  long tile = (long)blockIdx.x * 4 + wave;
  long g0 = tile * 16;

  // LayerNorm: 2 lanes per row (32 elems each)
  int r = lane >> 1, hh = lane & 1;
  const float* xr = x + (g0 + r) * WD + hh * 32;
  float xv[32];
  float s1 = 0.f, s2 = 0.f;
#pragma unroll
  for (int i = 0; i < 32; ++i) { float t = xr[i]; xv[i] = t; s1 += t; s2 += t * t; }
  s1 += __shfl_xor(s1, 1, 32);
  s2 += __shfl_xor(s2, 1, 32);
  float mean = s1 * (1.f / 64.f);
  float var = s2 * (1.f / 64.f) - mean * mean;
  float rstd = rsqrtf(var + 1e-5f);
  float dotwi = 0.f;
#pragma unroll
  for (int i = 0; i < 32; ++i) {
    int col = hh * 32 + i;
    float xn = (xv[i] - mean) * rstd * ln_g[col] + ln_b[col];
    lds[wave][r * 64 + col] = (h16)xn;
    dotwi += xn * wi_w[col];
  }
  dotwi += __shfl_xor(dotwi, 1, 32);
  if (hh == 0) logit_i[g0 + r] = dotwi + wi_b[0];
  __syncthreads();

  int m = lane & 15;
  v16h a0 = load_a16x32(&lds[wave][0], 64, m, hi, 0);
  v16h a1 = load_a16x32(&lds[wave][0], 64, m, hi, 32);

  v8f ckA[4], cvA[4];
#pragma unroll
  for (int jt = 0; jt < 4; ++jt) {
    int j = jt * 16 + m;
    v8f ck = {}; v8f cv = {};
#pragma unroll
    for (int ks = 0; ks < 2; ++ks) {
      int kb = ks * 32 + (hi ? 16 : 0);
      v16h bk, bv;
#pragma unroll
      for (int rr = 0; rr < 8; ++rr) {
        int kk = kb + 2 * rr;
        bk[2 * rr]     = (h16)Wk[j * WD + kk];
        bk[2 * rr + 1] = (h16)Wk[j * WD + kk + 1];
        bv[2 * rr]     = (h16)Wv[j * WD + kk];
        bv[2 * rr + 1] = (h16)Wv[j * WD + kk + 1];
      }
      v16h aa = ks ? a1 : a0;
      ck = wmma16(aa, bk, ck);
      cv = wmma16(aa, bv, cv);
    }
    ckA[jt] = ck; cvA[jt] = cv;
  }
  // row-wise l2 norm of k across the C-fragments (16-lane group reduction), then store f16
#pragma unroll
  for (int rr = 0; rr < 8; ++rr) {
    float p = 0.f;
#pragma unroll
    for (int jt = 0; jt < 4; ++jt) p += ckA[jt][rr] * ckA[jt][rr];
    p += __shfl_xor(p, 1, 32); p += __shfl_xor(p, 2, 32);
    p += __shfl_xor(p, 4, 32); p += __shfl_xor(p, 8, 32);
    float scale = 1.f / fmaxf(sqrtf(p), 1e-12f);
    int mrow = rr + (hi ? 8 : 0);
#pragma unroll
    for (int jt = 0; jt < 4; ++jt) {
      kn_h[(g0 + mrow) * WD + jt * 16 + m]  = (h16)(ckA[jt][rr] * scale);
      val_h[(g0 + mrow) * WD + jt * 16 + m] = (h16)(cvA[jt][rr]);
    }
  }
}

// ---------------- K2: per-batch LSE of wi logits ----------------
__global__ void k_lse_a(const float* __restrict__ logit_i, float* __restrict__ lse_a) {
  __shared__ float red[256];
  int b = blockIdx.x, tid = threadIdx.x;
  const float* lg = logit_i + (long)b * WN;
  float m = -1e30f;
  for (int n = tid; n < WN; n += 256) m = fmaxf(m, lg[n]);
  red[tid] = m; __syncthreads();
  for (int o = 128; o > 0; o >>= 1) { if (tid < o) red[tid] = fmaxf(red[tid], red[tid + o]); __syncthreads(); }
  m = red[0]; __syncthreads();
  float s = 0.f;
  for (int n = tid; n < WN; n += 256) s += expf(lg[n] - m);
  red[tid] = s; __syncthreads();
  for (int o = 128; o > 0; o >>= 1) { if (tid < o) red[tid] += red[tid + o]; __syncthreads(); }
  if (tid == 0) lse_a[b] = m + logf(red[0]);
}

// ---------------- K3: slots = mu + (|sigma|+eps)*noise ----------------
__global__ void k_slots_init(const float* __restrict__ mu, const float* __restrict__ sg,
                             const float* __restrict__ noise, float* __restrict__ slots) {
  int i = blockIdx.x * 256 + threadIdx.x;
  if (i < WB * WS * WD) {
    int d = i & 63;
    slots[i] = mu[d] + (fabsf(sg[d]) + CEPS) * noise[i];
  }
}

// ---------------- K4: slot LN + ws softmax (log_b) + q projection + l2norm ----------------
__global__ void k_slot_head(const float* __restrict__ slots,
                            const float* __restrict__ g, const float* __restrict__ bb,
                            const float* __restrict__ ws_w, const float* __restrict__ ws_b,
                            const float* __restrict__ Wq,
                            h16* __restrict__ qn_h, float* __restrict__ log_b) {
  __shared__ float sn[WS][WD];
  __shared__ float red[64];
  __shared__ float lg[8];
  int b = blockIdx.x, tid = threadIdx.x;
  for (int s = 0; s < WS; ++s) {
    float v = slots[((long)b * WS + s) * WD + tid];
    float mean = blockReduce64(v, red, tid) * (1.f / 64.f);
    float dv = v - mean;
    float var = blockReduce64(dv * dv, red, tid) * (1.f / 64.f);
    float xn = dv * rsqrtf(var + 1e-5f) * g[tid] + bb[tid];
    sn[s][tid] = xn;
    float lp = blockReduce64(xn * ws_w[tid], red, tid);
    if (tid == 0) lg[s] = lp + ws_b[0];
    __syncthreads();
  }
  if (tid == 0) {
    float m = lg[0];
    for (int s = 1; s < WS; ++s) m = fmaxf(m, lg[s]);
    float ss = 0.f;
    for (int s = 0; s < WS; ++s) ss += expf(lg[s] - m);
    lg[7] = m + logf(ss);
  }
  __syncthreads();
  if (tid < WS) log_b[b * WS + tid] = lg[tid] - lg[7] + LOG_S;
  for (int s = 0; s < WS; ++s) {
    float q = 0.f;
    for (int k = 0; k < WD; ++k) q += sn[s][k] * Wq[tid * WD + k];
    float nrm = blockReduce64(q * q, red, tid);
    float scale = 1.f / fmaxf(sqrtf(nrm), 1e-12f);
    qn_h[((long)b * 16 + s) * WD + tid] = (h16)(q * scale);
  }
  for (int s = WS; s < 16; ++s) qn_h[((long)b * 16 + s) * WD + tid] = (h16)0.f;
}

// ---------------- K5: C = 1 - kn . qn^T (WMMA) ----------------
__global__ void k_cost(const h16* __restrict__ kn_h, const h16* __restrict__ qn_h,
                       float* __restrict__ C) {
  int wave = threadIdx.x >> 5, lane = threadIdx.x & 31;
  bool hi = lane >= 16; int m = lane & 15;
  long tile = (long)blockIdx.x * 4 + wave;
  long g0 = tile * 16;
  int b = (int)(g0 >> 14);
  v8f acc = {};
#pragma unroll
  for (int ks = 0; ks < 2; ++ks) {
    v16h a = load_a16x32(kn_h + g0 * WD, WD, m, hi, ks * 32);
    int kb = ks * 32 + (hi ? 16 : 0);
    v16h bq;
#pragma unroll
    for (int rr = 0; rr < 8; ++rr) {
      bq[2 * rr]     = qn_h[((long)b * 16 + m) * WD + kb + 2 * rr];
      bq[2 * rr + 1] = qn_h[((long)b * 16 + m) * WD + kb + 2 * rr + 1];
    }
    acc = wmma16(a, bq, acc);
  }
  if (m < WS) {
#pragma unroll
    for (int rr = 0; rr < 8; ++rr) {
      int mr = rr + (hi ? 8 : 0);
      C[(g0 + mr) * WS + m] = 1.0f - acc[rr];
    }
  }
}

// ---------------- K6: u = log_a - LSE_s(-C + v) ----------------
__global__ void k_u(const float* __restrict__ C, const float* __restrict__ vin,
                    const float* __restrict__ logit_i, const float* __restrict__ lse_a,
                    float* __restrict__ uout) {
  long row = (long)blockIdx.x * 256 + threadIdx.x;
  int b = (int)(row >> 14);
  const float* c = C + row * WS;
  float t[WS]; float m = -1e30f;
#pragma unroll
  for (int s = 0; s < WS; ++s) { t[s] = -c[s] + vin[b * WS + s]; m = fmaxf(m, t[s]); }
  float sum = 0.f;
#pragma unroll
  for (int s = 0; s < WS; ++s) sum += expf(t[s] - m);
  float lse = m + logf(sum);
  float la = logit_i[row] - lse_a[b] + LOG_S;
  uout[row] = la - lse;
}

// ---------------- K7: v = log_b - LSE_n(-C + u) ----------------
__global__ void k_v(const float* __restrict__ C, const float* __restrict__ u,
                    const float* __restrict__ log_b, float* __restrict__ vout) {
  __shared__ float red[256];
  int blk = blockIdx.x; int b = blk / WS, s = blk % WS;
  int tid = threadIdx.x;
  const float* cb = C + (long)b * WN * WS + s;
  const float* ub = u + (long)b * WN;
  float m = -1e30f;
  for (int n = tid; n < WN; n += 256) m = fmaxf(m, -cb[(long)n * WS] + ub[n]);
  red[tid] = m; __syncthreads();
  for (int o = 128; o > 0; o >>= 1) { if (tid < o) red[tid] = fmaxf(red[tid], red[tid + o]); __syncthreads(); }
  m = red[0]; __syncthreads();
  float sum = 0.f;
  for (int n = tid; n < WN; n += 256) sum += expf(-cb[(long)n * WS] + ub[n] - m);
  red[tid] = sum; __syncthreads();
  for (int o = 128; o > 0; o >>= 1) { if (tid < o) red[tid] += red[tid + o]; __syncthreads(); }
  if (tid == 0) vout[blk] = log_b[blk] - (m + logf(red[0]));
}

// ---------------- K8: entropy gradient init: dP = attn*dA, du5, dv5-partials ----------------
__global__ void k_grad_init(const float* __restrict__ C, const float* __restrict__ u5,
                            const float* __restrict__ v5,
                            float* __restrict__ dP, float* __restrict__ du5,
                            float* __restrict__ partials) {
  __shared__ float red[256];
  long row = (long)blockIdx.x * 256 + threadIdx.x;
  int b = (int)(row >> 14);
  int tid = threadIdx.x;
  const float* c = C + row * WS;
  float uu = u5[row];
  float dsum = 0.f;
  float dp[WS];
#pragma unroll
  for (int s = 0; s < WS; ++s) {
    float a = expf(-c[s] + uu + v5[b * WS + s]);
    float dA = -(logf(a + CEPS) + a / (a + CEPS));
    dp[s] = a * dA;
    dsum += dp[s];
    dP[row * WS + s] = dp[s];
  }
  du5[row] = dsum;
  for (int s = 0; s < WS; ++s) {
    float p = blockReduce256(dp[s], red, tid);
    if (tid == 0) partials[(long)blockIdx.x * WS + s] = p;
  }
}

// ---------------- K9: fused backward level t (and C update at t==1) ----------------
// flags bit0: add du5 init term (t==N_SH); bit1: t==1 -> update C, skip dv_out partials
__global__ void k_bwd(float* __restrict__ C, float* __restrict__ dP,
                      const float* __restrict__ ut, const float* __restrict__ vt,
                      const float* __restrict__ vprev,
                      const float* __restrict__ dv_in, float* __restrict__ partials,
                      const float* __restrict__ du5,
                      const float* __restrict__ logit_i, const float* __restrict__ lse_a,
                      const float* __restrict__ log_b, int flags) {
  __shared__ float red[256];
  long row = (long)blockIdx.x * 256 + threadIdx.x;
  int b = (int)(row >> 14);
  int tid = threadIdx.x;
  float* c = C + row * WS;
  float uu = ut[row];
  float la = logit_i[row] - lse_a[b] + LOG_S;
  float w[WS], w2[WS];
  float du = (flags & 1) ? du5[row] : 0.f;
#pragma unroll
  for (int s = 0; s < WS; ++s) {
    w[s]  = expf(-c[s] + uu - log_b[b * WS + s] + vt[b * WS + s]);
    w2[s] = expf(-c[s] + vprev[b * WS + s] - la + uu);
    du -= w[s] * dv_in[b * WS + s];
  }
  float dvp[WS];
#pragma unroll
  for (int s = 0; s < WS; ++s) {
    float d = dP[row * WS + s] - w[s] * dv_in[b * WS + s] - w2[s] * du;
    dP[row * WS + s] = d;
    dvp[s] = -w2[s] * du;
    if (flags & 2) c[s] = c[s] + d;   // C <- C - lr * dC, dC = -dP, lr = 1
  }
  if (!(flags & 2)) {
    for (int s = 0; s < WS; ++s) {
      float p = blockReduce256(dvp[s], red, tid);
      if (tid == 0) partials[(long)blockIdx.x * WS + s] = p;
    }
  }
}

// deterministic second-stage reduce of per-block partials -> dv (B,S)
__global__ void k_dv_finish(const float* __restrict__ partials, float* __restrict__ dv) {
  int i = threadIdx.x;
  if (i < WB * WS) {
    int b = i / WS, s = i % WS;
    float sum = 0.f;
    for (int blk = 0; blk < 64; ++blk) sum += partials[((long)(b * 64 + blk)) * WS + s];
    dv[i] = sum;
  }
}

// ---------------- K10: attn_t = exp(-C + u + v), transposed write ----------------
__global__ void k_attn(const float* __restrict__ C, const float* __restrict__ u5,
                       const float* __restrict__ v5, float* __restrict__ attn_t) {
  long row = (long)blockIdx.x * 256 + threadIdx.x;
  int b = (int)(row >> 14);
  int n = (int)(row & (WN - 1));
  float uu = u5[row];
#pragma unroll
  for (int s = 0; s < WS; ++s) {
    float a = expf(-C[row * WS + s] + uu + v5[b * WS + s]);
    attn_t[((long)b * WS + s) * WN + n] = a;
  }
}

// ---------------- K11: updates = attn_t @ val (WMMA, split-K over 8 waves) ----------------
__global__ void k_updates(const float* __restrict__ attn_t, const h16* __restrict__ val_h,
                          float* __restrict__ updates) {
  __shared__ float red[8][16][WD];
  int b = blockIdx.x;
  int wave = threadIdx.x >> 5, lane = threadIdx.x & 31;
  bool hi = lane >= 16; int m = lane & 15;
  v8f acc[4] = {{}, {}, {}, {}};
  for (int ks = 0; ks < 64; ++ks) {
    int nb = wave * 2048 + ks * 32;
    v16h a;
#pragma unroll
    for (int rr = 0; rr < 8; ++rr) {
      int k0 = (hi ? 16 : 0) + 2 * rr;
      if (m < WS) {
        const float* ap = attn_t + ((long)b * WS + m) * WN + nb + k0;
        a[2 * rr]     = (h16)ap[0];
        a[2 * rr + 1] = (h16)ap[1];
      } else { a[2 * rr] = (h16)0.f; a[2 * rr + 1] = (h16)0.f; }
    }
#pragma unroll
    for (int jt = 0; jt < 4; ++jt) {
      int j = jt * 16 + m;
      int kb = nb + (hi ? 16 : 0);
      v16h bv;
#pragma unroll
      for (int rr = 0; rr < 8; ++rr) {
        bv[2 * rr]     = val_h[((long)b * WN + kb + 2 * rr) * WD + j];
        bv[2 * rr + 1] = val_h[((long)b * WN + kb + 2 * rr + 1) * WD + j];
      }
      acc[jt] = wmma16(a, bv, acc[jt]);
    }
  }
#pragma unroll
  for (int rr = 0; rr < 8; ++rr) {
    int mr = rr + (hi ? 8 : 0);
#pragma unroll
    for (int jt = 0; jt < 4; ++jt) red[wave][mr][jt * 16 + m] = acc[jt][rr];
  }
  __syncthreads();
  for (int i = threadIdx.x; i < WS * WD; i += 256) {
    int s = i / WD, d = i % WD;
    float sum = 0.f;
#pragma unroll
    for (int w = 0; w < 8; ++w) sum += red[w][s][d];
    updates[((long)b * WS + s) * WD + d] = sum;
  }
}

// ---------------- K12: GRUCell + LN + MLP residual per slot ----------------
__global__ void k_gru_ff(const float* __restrict__ updates, float* __restrict__ slots,
                         const float* __restrict__ w_ih, const float* __restrict__ w_hh,
                         const float* __restrict__ b_ih, const float* __restrict__ b_hh,
                         const float* __restrict__ fc1_w, const float* __restrict__ fc1_b,
                         const float* __restrict__ fc2_w, const float* __restrict__ fc2_b,
                         const float* __restrict__ g, const float* __restrict__ bb) {
  __shared__ float up[WD], sp[WD], fn[WD], hbuf[WH], red[64];
  long row = blockIdx.x;
  int tid = threadIdx.x;
  up[tid] = updates[row * WD + tid];
  sp[tid] = slots[row * WD + tid];
  __syncthreads();
  float xr = b_ih[tid], xz = b_ih[64 + tid], xn = b_ih[128 + tid];
  float hr = b_hh[tid], hz = b_hh[64 + tid], hn = b_hh[128 + tid];
  for (int k = 0; k < WD; ++k) {
    float uk = up[k], sk = sp[k];
    xr += uk * w_ih[(long)tid * WD + k];
    xz += uk * w_ih[(long)(64 + tid) * WD + k];
    xn += uk * w_ih[(long)(128 + tid) * WD + k];
    hr += sk * w_hh[(long)tid * WD + k];
    hz += sk * w_hh[(long)(64 + tid) * WD + k];
    hn += sk * w_hh[(long)(128 + tid) * WD + k];
  }
  float r = 1.f / (1.f + expf(-(xr + hr)));
  float z = 1.f / (1.f + expf(-(xz + hz)));
  float nn = tanhf(xn + r * hn);
  float h = (1.f - z) * nn + z * sp[tid];
  float mean = blockReduce64(h, red, tid) * (1.f / 64.f);
  float dv = h - mean;
  float var = blockReduce64(dv * dv, red, tid) * (1.f / 64.f);
  fn[tid] = dv * rsqrtf(var + 1e-5f) * g[tid] + bb[tid];
  __syncthreads();
#pragma unroll
  for (int oo = 0; oo < 2; ++oo) {
    int o = tid + oo * 64;
    float acc = fc1_b[o];
    for (int k = 0; k < WD; ++k) acc += fn[k] * fc1_w[(long)o * WD + k];
    hbuf[o] = fmaxf(acc, 0.f);
  }
  __syncthreads();
  float out = h + fc2_b[tid];
  for (int k = 0; k < WH; ++k) out += hbuf[k] * fc2_w[(long)tid * WH + k];
  slots[row * WD + tid] = out;
}

__global__ void k_zero(float* p, int n) { int i = blockIdx.x * 256 + threadIdx.x; if (i < n) p[i] = 0.f; }
__global__ void k_copy(float* dst, const float* src, int n) { int i = blockIdx.x * 256 + threadIdx.x; if (i < n) dst[i] = src[i]; }

extern "C" void kernel_launch(void* const* d_in, const int* in_sizes, int n_in,
                              void* d_out, int out_size, void* d_ws, size_t ws_size,
                              hipStream_t stream) {
  (void)in_sizes; (void)n_in; (void)out_size; (void)ws_size;
  const float* inputs      = (const float*)d_in[0];
  const float* slot_noise  = (const float*)d_in[1];
  const float* slots_mu    = (const float*)d_in[2];
  const float* slots_sigma = (const float*)d_in[3];
  const float* Wq = (const float*)d_in[4];
  const float* Wk = (const float*)d_in[5];
  const float* Wv = (const float*)d_in[6];
  const float* gru_w_ih = (const float*)d_in[7];
  const float* gru_w_hh = (const float*)d_in[8];
  const float* gru_b_ih = (const float*)d_in[9];
  const float* gru_b_hh = (const float*)d_in[10];
  const float* fc1_w = (const float*)d_in[11];
  const float* fc1_b = (const float*)d_in[12];
  const float* fc2_w = (const float*)d_in[13];
  const float* fc2_b = (const float*)d_in[14];
  const float* ln_in_g = (const float*)d_in[15];
  const float* ln_in_b = (const float*)d_in[16];
  const float* ln_sl_g = (const float*)d_in[17];
  const float* ln_sl_b = (const float*)d_in[18];
  const float* ln_ff_g = (const float*)d_in[19];
  const float* ln_ff_b = (const float*)d_in[20];
  const float* wi_w = (const float*)d_in[21];
  const float* wi_b = (const float*)d_in[22];
  const float* ws_w = (const float*)d_in[23];
  const float* ws_b = (const float*)d_in[24];

  char* p = (char*)d_ws;
  auto alloc = [&](size_t bytes) -> char* {
    char* r = p; p += (bytes + 255) & ~(size_t)255; return r;
  };
  h16* kn_h  = (h16*)alloc(sizeof(h16) * (size_t)WB * WN * WD);
  h16* val_h = (h16*)alloc(sizeof(h16) * (size_t)WB * WN * WD);
  float* logit_i = (float*)alloc(4ull * WB * WN);
  float* lse_a   = (float*)alloc(4ull * WB);
  float* C  = (float*)alloc(4ull * WB * WN * WS);
  float* dP = (float*)alloc(4ull * WB * WN * WS);
  float* U[6]; U[0] = nullptr;
  for (int t = 1; t <= 5; ++t) U[t] = (float*)alloc(4ull * WB * WN);
  float* V[6];
  for (int t = 0; t <= 5; ++t) V[t] = (float*)alloc(4ull * WB * WS);
  float* du5 = (float*)alloc(4ull * WB * WN);
  float* dvA = (float*)alloc(4ull * WB * WS);
  float* dvB = (float*)alloc(4ull * WB * WS);
  float* partials = (float*)alloc(4ull * (WB * WN / 256) * WS);
  h16* qn_h = (h16*)alloc(sizeof(h16) * (size_t)WB * 16 * WD);
  float* log_b = (float*)alloc(4ull * WB * WS);
  float* slots = (float*)alloc(4ull * WB * WS * WD);
  float* updates = (float*)alloc(4ull * WB * WS * WD);
  float* attn_t = (float*)d_out + (size_t)WB * WS * WD;   // output region 2

  const int ROWB = WB * WN / 256;   // 1024 blocks of 256 rows

  k_ln_proj<<<WB * WN / 64, 128, 0, stream>>>(inputs, ln_in_g, ln_in_b, Wk, Wv, wi_w, wi_b,
                                              kn_h, val_h, logit_i);
  k_lse_a<<<WB, 256, 0, stream>>>(logit_i, lse_a);
  k_slots_init<<<(WB * WS * WD + 255) / 256, 256, 0, stream>>>(slots_mu, slots_sigma, slot_noise, slots);

  for (int it = 0; it < N_ITERS; ++it) {
    k_slot_head<<<WB, 64, 0, stream>>>(slots, ln_sl_g, ln_sl_b, ws_w, ws_b, Wq, qn_h, log_b);
    k_cost<<<WB * WN / 64, 128, 0, stream>>>(kn_h, qn_h, C);
    k_zero<<<1, 128, 0, stream>>>(V[0], WB * WS);        // v warm start = 0 per _mesh call
    for (int mm = 0; mm < N_MESH; ++mm) {
      // forward Sinkhorn, storing all duals for the analytic backward
      for (int t = 1; t <= N_SH; ++t) {
        k_u<<<ROWB, 256, 0, stream>>>(C, V[t - 1], logit_i, lse_a, U[t]);
        k_v<<<WB * WS, 256, 0, stream>>>(C, U[t], log_b, V[t]);
      }
      // backward through entropy of the plan
      k_grad_init<<<ROWB, 256, 0, stream>>>(C, U[5], V[5], dP, du5, partials);
      k_dv_finish<<<1, 128, 0, stream>>>(partials, dvA);
      float* din = dvA; float* dnx = dvB;
      for (int t = N_SH; t >= 1; --t) {
        int flags = (t == N_SH ? 1 : 0) | (t == 1 ? 2 : 0);
        k_bwd<<<ROWB, 256, 0, stream>>>(C, dP, U[t], V[t], V[t - 1], din, partials, du5,
                                        logit_i, lse_a, log_b, flags);
        if (t > 1) {
          k_dv_finish<<<1, 128, 0, stream>>>(partials, dnx);
          float* tmp = din; din = dnx; dnx = tmp;
        }
      }
      // warm-start duals for next MESH step come from this step's aux (pre-update C)
      k_copy<<<1, 128, 0, stream>>>(V[0], V[5], WB * WS);
    }
    // final Sinkhorn with warm-started v
    for (int t = 1; t <= N_SH; ++t) {
      k_u<<<ROWB, 256, 0, stream>>>(C, V[t - 1], logit_i, lse_a, U[t]);
      k_v<<<WB * WS, 256, 0, stream>>>(C, U[t], log_b, V[t]);
    }
    k_attn<<<ROWB, 256, 0, stream>>>(C, U[5], V[5], attn_t);
    k_updates<<<WB, 256, 0, stream>>>(attn_t, val_h, updates);
    k_gru_ff<<<WB * WS, 64, 0, stream>>>(updates, slots, gru_w_ih, gru_w_hh, gru_b_ih, gru_b_hh,
                                         fc1_w, fc1_b, fc2_w, fc2_b, ln_ff_g, ln_ff_b);
  }
  k_copy<<<(WB * WS * WD + 255) / 256, 256, 0, stream>>>((float*)d_out, slots, WB * WS * WD);
}
